// SparseMoENetwork_27341761806751
// MI455X (gfx1250) — matrix operand order repaired and checked by reading, not compiled, not run on hardware
//
#include <hip/hip_runtime.h>
#include <hip/hip_bf16.h>
#include <math.h>

// ---------------- problem constants ----------------
#define B_N     2048
#define IN_DIM  512
#define T_N     50
#define XROW    562          // IN_DIM + T_N
#define E_N     16
#define K_TOP   4
#define W_DIM   1024
#define HD_DIM  256
#define NEXP    18           // 16 routed + 2 shared pseudo-experts
// padded routed capacity: 8192 assignments + 16*63 pad <= 9216
#define SH0     9216
#define SH1     11264
#define CAP     13312        // SH1 + 2048

// LDS anti-bank-conflict padding (elements = bf16)
#define A_STRIDE  40                  // 32 data + 8 pad  -> 20-dword lane stride
#define B_LSTRIDE 24                  // 16 data + 8 pad  -> 12-dword lane stride, 16B aligned
#define B_FRAG    (32 * B_LSTRIDE)    // 768 elements per fragment in LDS
#define NFRAG     16                  // B fragments per block step (BN = 256)

typedef __bf16  bf16_t;
typedef __attribute__((ext_vector_type(16))) bf16_t v16bf;
typedef __attribute__((ext_vector_type(8)))  float  v8f;

union Frag { uint4 u[2]; v16bf v; };

__device__ __forceinline__ unsigned short f2bf(float f) {
    union { float f; unsigned u; } c; c.f = f;
    unsigned r = (c.u + 0x7FFFu + ((c.u >> 16) & 1u)) >> 16;   // RNE
    return (unsigned short)r;
}

// ---------------- 1) gating: task argmax, softmax, top-4 ----------------
__global__ void gating_kernel(const float* __restrict__ x,
                              const float* __restrict__ Wg,
                              const float* __restrict__ gbias,
                              int* __restrict__ counts,
                              int* __restrict__ task_of,
                              int* __restrict__ topk_e,
                              float* __restrict__ topk_w) {
    int b = blockIdx.x * blockDim.x + threadIdx.x;
    if (b >= B_N) return;
    const float* row = x + (size_t)b * XROW;
    int t = 0; float bm = row[IN_DIM];
    for (int i = 1; i < T_N; ++i) { float v = row[IN_DIM + i]; if (v > bm) { bm = v; t = i; } }
    task_of[b] = t;
    float lg[E_N], key[E_N];
    float m = -1e30f;
    for (int e = 0; e < E_N; ++e) { lg[e] = Wg[t * E_N + e]; m = fmaxf(m, lg[e]); }
    float den = 0.f;
    for (int e = 0; e < E_N; ++e) { key[e] = __expf(lg[e] - m); den += key[e]; }
    float inv = 1.f / den;
    for (int e = 0; e < E_N; ++e) key[e] = key[e] * inv + gbias[e];   // score + bias
    int idx[K_TOP];
    for (int k = 0; k < K_TOP; ++k) {
        float best = -1e30f; int bi = 0;
        for (int e = 0; e < E_N; ++e) if (key[e] > best) { best = key[e]; bi = e; }
        idx[k] = bi; key[bi] = -3.0e38f;
    }
    float mm = lg[idx[0]];
    for (int k = 1; k < K_TOP; ++k) mm = fmaxf(mm, lg[idx[k]]);
    float wv[K_TOP], ss = 0.f;
    for (int k = 0; k < K_TOP; ++k) { wv[k] = __expf(lg[idx[k]] - mm); ss += wv[k]; }
    float is = 1.f / ss;
    for (int k = 0; k < K_TOP; ++k) {
        topk_e[b * K_TOP + k] = idx[k];
        topk_w[b * K_TOP + k] = wv[k] * is;
        atomicAdd(&counts[idx[k]], 1);
    }
}

// ---------------- 2) build 64-padded expert segments ----------------
__global__ void build_segs(const int* __restrict__ counts,
                           int* __restrict__ segs /* lo[18], hi[18] */,
                           int* __restrict__ cursor) {
    if (threadIdx.x == 0 && blockIdx.x == 0) {
        int acc = 0;
        for (int e = 0; e < E_N; ++e) {
            segs[e] = acc; cursor[e] = acc;
            acc += (counts[e] + 63) & ~63;
            segs[NEXP + e] = acc;
        }
        segs[16] = SH0; segs[NEXP + 16] = SH1;
        segs[17] = SH1; segs[NEXP + 17] = CAP;
    }
}

// ---------------- 3) scatter tokens into expert segments (bf16 gather) ----------------
__global__ void scatter_gather(const float* __restrict__ x,
                               const int* __restrict__ topk_e,
                               int* __restrict__ cursor,
                               int* __restrict__ slot_of,
                               unsigned short* __restrict__ Ag) {
    int b = blockIdx.x;
    __shared__ int slots[K_TOP];
    if (threadIdx.x < K_TOP) {
        int e = topk_e[b * K_TOP + threadIdx.x];
        int s = atomicAdd(&cursor[e], 1);
        slots[threadIdx.x] = s;
        slot_of[b * K_TOP + threadIdx.x] = s;
    }
    __syncthreads();
    const float* row = x + (size_t)b * XROW;
    int i0 = threadIdx.x, i1 = threadIdx.x + 256;
    unsigned short v0 = f2bf(row[i0]);
    unsigned short v1 = f2bf(row[i1]);
    int dst6[6] = { slots[0], slots[1], slots[2], slots[3], SH0 + b, SH1 + b };
    for (int d = 0; d < 6; ++d) {
        unsigned short* dst = Ag + (size_t)dst6[d] * IN_DIM;
        dst[i0] = v0; dst[i1] = v1;
    }
}

// ---------------- 4) repack f32 weights -> bf16 in WMMA fragment layout ----------------
// Wp[e][kb][nt][lane(32)][16]: lane holds column n = nt*16+(lane&15),
// K = kb*32 + (lane>>4)*8 + (j&7) + (j>>3)*16  (matches V_WMMA 16-bit B-operand layout).
__global__ void repack_w(const float* __restrict__ Wr,  // [16][Kdim][1024]
                         const float* __restrict__ Ws,  // [2][Kdim][1024]
                         unsigned short* __restrict__ Wp,
                         int Kdim) {
    int KB  = Kdim >> 5;
    int blk = blockIdx.x;
    int sub = blk & 7;  int ekb = blk >> 3;
    int kb  = ekb % KB; int e   = ekb / KB;
    const float* W = (e < E_N) ? (Wr + (size_t)e * Kdim * W_DIM)
                               : (Ws + (size_t)(e - E_N) * Kdim * W_DIM);
    int p    = sub * 256 + threadIdx.x;  // 2048 = 64 nt * 32 lanes
    int nt   = p >> 5;
    int lane = p & 31;
    int n     = nt * 16 + (lane & 15);
    int kbase = kb * 32 + (lane >> 4) * 8;
    union { unsigned short s[16]; uint4 u[2]; } tmp;
    #pragma unroll
    for (int j = 0; j < 16; ++j)
        tmp.s[j] = f2bf(W[(size_t)(kbase + (j & 7) + (j >> 3) * 16) * W_DIM + n]);
    unsigned short* dst = Wp + (((size_t)(e * KB + kb) * 64 + nt) * 512 + lane * 16);
    ((uint4*)dst)[0] = tmp.u[0];
    ((uint4*)dst)[1] = tmp.u[1];
}

// ---------------- 5) segment-aware WMMA GEMM: Out = act(A @ W[e] + b[e]) ----------------
// Block tile 64(M) x 256(N) x 32(K), 256 threads = 8 wave32 waves (2 M-bands x 4 N-groups).
// Each wave owns a 32x64 C tile: 8 accumulators, outer-product reuse of a0/a1 x b0..b3
// -> 8 x v_wmma_f32_16x16x32_bf16 per 12 ds_load_b128. LDS double-buffered, register
// prefetch of tile k+1 overlaps global latency with WMMA of tile k; 1 barrier/step.
template <bool RELU, bool OUTF32>
__global__ __launch_bounds__(256)
void moe_gemm(const unsigned short* __restrict__ A, int ldk, int KB,
              const unsigned short* __restrict__ Wp,
              const float* __restrict__ bias_r,
              const float* __restrict__ bias_s,
              void* __restrict__ Out,
              const int* __restrict__ segs) {
    int row0 = blockIdx.x * 64;
    int e = -1;
    for (int i = 0; i < NEXP; ++i)
        if (row0 >= segs[i] && row0 < segs[NEXP + i]) { e = i; break; }
    if (e < 0) return;                                // gap tile -> no work

    const float* bias = (e < E_N) ? (bias_r + e * W_DIM) : (bias_s + (e - E_N) * W_DIM);
    int n0  = blockIdx.y * 256;
    int nt0 = blockIdx.y * NFRAG;

    __shared__ __align__(16) unsigned short As[2][64 * A_STRIDE];     // 2 x 5 KB
    __shared__ __align__(16) unsigned short Bs[2][NFRAG * B_FRAG];    // 2 x 24 KB

    int lane = threadIdx.x & 31;
    int wid  = threadIdx.x >> 5;
    int wm   = wid & 1;          // M band (32 rows)
    int wn   = wid >> 1;         // N group (64 cols = 4 fragments)

    v8f acc[2][4];
    #pragma unroll
    for (int s = 0; s < 2; ++s)
        #pragma unroll
        for (int j = 0; j < 4; ++j) acc[s][j] = (v8f){};

    // A staging: one 16B chunk / thread (64x32 bf16 = 4 KB)
    int arow = threadIdx.x >> 2, aseg = threadIdx.x & 3;
    const size_t agbase = (size_t)(row0 + arow) * ldk + aseg * 8;
    const int    alds   = arow * A_STRIDE + aseg * 8;
    // B staging: four 16B chunks / thread (16 fragments = 16 KB), padded LDS layout
    int  bci[4], blds[4];
    #pragma unroll
    for (int q = 0; q < 4; ++q) {
        int c = threadIdx.x + 256 * q;
        bci[q]  = c;
        blds[q] = (c >> 5) * B_FRAG + ((c & 31) >> 1) * B_LSTRIDE + (c & 1) * 8;
    }
    const uint4* wsrc_all = (const uint4*)Wp;
    const size_t wbase = ((size_t)e * KB * 64 + nt0) * 64;   // uint4 units; +kb*4096 per step

    // fragment read offsets
    const int ar0 = (wm * 32 + (lane & 15)) * A_STRIDE + ((lane >> 4) * 8);
    const int ar1 = ar0 + 16 * A_STRIDE;
    int bf[4];
    #pragma unroll
    for (int j = 0; j < 4; ++j) bf[j] = (wn * 4 + j) * B_FRAG + lane * B_LSTRIDE;

    // ---- prologue: stage tile 0 into buffer 0 ----
    {
        uint4 pa = *(const uint4*)(A + agbase);
        uint4 pb[4];
        #pragma unroll
        for (int q = 0; q < 4; ++q) pb[q] = wsrc_all[wbase + bci[q]];
        *(uint4*)&As[0][alds] = pa;
        #pragma unroll
        for (int q = 0; q < 4; ++q) *(uint4*)&Bs[0][blds[q]] = pb[q];
    }
    __syncthreads();

    for (int kb = 0; kb + 1 < KB; ++kb) {
        int cur = kb & 1, nxt = cur ^ 1;
        // prefetch tile kb+1 into registers (global latency hidden behind WMMA)
        uint4 pa = *(const uint4*)(A + agbase + (size_t)(kb + 1) * 32);
        uint4 pb[4];
        #pragma unroll
        for (int q = 0; q < 4; ++q)
            pb[q] = wsrc_all[wbase + (size_t)(kb + 1) * 4096 + bci[q]];

        // compute tile kb from LDS buffer `cur`
        Frag a0, a1, b[4];
        a0.u[0] = *(const uint4*)&As[cur][ar0];
        a0.u[1] = *(const uint4*)&As[cur][ar0 + 16];
        a1.u[0] = *(const uint4*)&As[cur][ar1];
        a1.u[1] = *(const uint4*)&As[cur][ar1 + 16];
        #pragma unroll
        for (int j = 0; j < 4; ++j) {
            b[j].u[0] = *(const uint4*)&Bs[cur][bf[j]];
            b[j].u[1] = *(const uint4*)&Bs[cur][bf[j] + 8];
        }
        #pragma unroll
        for (int j = 0; j < 4; ++j) {
            acc[0][j] = __builtin_amdgcn_wmma_f32_16x16x32_bf16(false, a0.v, false, b[j].v, (short)0, acc[0][j], false, false);
            acc[1][j] = __builtin_amdgcn_wmma_f32_16x16x32_bf16(false, a1.v, false, b[j].v, (short)0, acc[1][j], false, false);
        }

        // commit prefetched tile to the other buffer
        *(uint4*)&As[nxt][alds] = pa;
        #pragma unroll
        for (int q = 0; q < 4; ++q) *(uint4*)&Bs[nxt][blds[q]] = pb[q];
        __syncthreads();
    }

    // ---- peeled last tile ----
    {
        int cur = (KB - 1) & 1;
        Frag a0, a1, b[4];
        a0.u[0] = *(const uint4*)&As[cur][ar0];
        a0.u[1] = *(const uint4*)&As[cur][ar0 + 16];
        a1.u[0] = *(const uint4*)&As[cur][ar1];
        a1.u[1] = *(const uint4*)&As[cur][ar1 + 16];
        #pragma unroll
        for (int j = 0; j < 4; ++j) {
            b[j].u[0] = *(const uint4*)&Bs[cur][bf[j]];
            b[j].u[1] = *(const uint4*)&Bs[cur][bf[j] + 8];
        }
        #pragma unroll
        for (int j = 0; j < 4; ++j) {
            acc[0][j] = __builtin_amdgcn_wmma_f32_16x16x32_bf16(false, a0.v, false, b[j].v, (short)0, acc[0][j], false, false);
            acc[1][j] = __builtin_amdgcn_wmma_f32_16x16x32_bf16(false, a1.v, false, b[j].v, (short)0, acc[1][j], false, false);
        }
    }

    // epilogue: C layout — VGPR r: lanes 0-15 -> M=r, lanes 16-31 -> M=r+8; N = lane&15
    int colL = lane & 15, hi = lane >> 4;
    #pragma unroll
    for (int sub = 0; sub < 2; ++sub) {
        #pragma unroll
        for (int j = 0; j < 4; ++j) {
            v8f c = acc[sub][j];
            int n = n0 + (wn * 4 + j) * 16 + colL;
            float bv = bias[n];
            #pragma unroll
            for (int r = 0; r < 8; ++r) {
                float val = c[r] + bv;
                if (RELU) val = fmaxf(val, 0.f);
                size_t rowg = (size_t)(row0 + wm * 32 + sub * 16 + hi * 8 + r);
                if (OUTF32) ((float*)Out)[rowg * W_DIM + n] = val;
                else        ((unsigned short*)Out)[rowg * W_DIM + n] = f2bf(val);
            }
        }
    }
}

// ---------------- 6) weighted routed + mean shared, tanh ----------------
__global__ void combine_kernel(const float* __restrict__ out3,
                               const int* __restrict__ slot_of,
                               const float* __restrict__ topk_w,
                               float* __restrict__ fout) {
    int b = blockIdx.x;
    int w = blockIdx.y * 256 + threadIdx.x;
    float acc = 0.f;
    #pragma unroll
    for (int k = 0; k < K_TOP; ++k) {
        int s = slot_of[b * K_TOP + k];
        acc += topk_w[b * K_TOP + k] * out3[(size_t)s * W_DIM + w];
    }
    acc += 0.5f * (out3[(size_t)(SH0 + b) * W_DIM + w] + out3[(size_t)(SH1 + b) * W_DIM + w]);
    fout[(size_t)b * W_DIM + w] = tanhf(acc);
}

// ---------------- 7) per-task head: out[b] = f[b] @ Wh[task] + bh[task] ----------------
__global__ void heads_kernel(const float* __restrict__ fout,
                             const float* __restrict__ Wh,
                             const float* __restrict__ bh,
                             const int* __restrict__ task_of,
                             float* __restrict__ out) {
    int b = blockIdx.x;
    int h = threadIdx.x;
    __shared__ float fsh[W_DIM];
    for (int i = threadIdx.x; i < W_DIM; i += 256) fsh[i] = fout[(size_t)b * W_DIM + i];
    __syncthreads();
    int t = task_of[b];
    const float* wp = Wh + (size_t)t * W_DIM * HD_DIM + h;
    float acc = bh[t * HD_DIM + h];
    #pragma unroll 4
    for (int i = 0; i < W_DIM; ++i) acc = fmaf(fsh[i], wp[(size_t)i * HD_DIM], acc);
    out[(size_t)b * HD_DIM + h] = acc;
}

// ---------------- host launcher ----------------
extern "C" void kernel_launch(void* const* d_in, const int* in_sizes, int n_in,
                              void* d_out, int out_size, void* d_ws, size_t ws_size,
                              hipStream_t stream) {
    const float* x     = (const float*)d_in[0];
    const float* Wg    = (const float*)d_in[1];
    const float* gbias = (const float*)d_in[2];
    const float* We1   = (const float*)d_in[3];
    const float* be1   = (const float*)d_in[4];
    const float* We2   = (const float*)d_in[5];
    const float* be2   = (const float*)d_in[6];
    const float* We3   = (const float*)d_in[7];
    const float* be3   = (const float*)d_in[8];
    const float* Ws1   = (const float*)d_in[9];
    const float* bs1   = (const float*)d_in[10];
    const float* Ws2   = (const float*)d_in[11];
    const float* bs2   = (const float*)d_in[12];
    const float* Ws3   = (const float*)d_in[13];
    const float* bs3   = (const float*)d_in[14];
    const float* Wh    = (const float*)d_in[15];
    const float* bh    = (const float*)d_in[16];

    // workspace layout
    char* ws = (char*)d_ws;
    size_t off = 0;
    auto alloc = [&](size_t bytes) { size_t o = off; off = (off + bytes + 255) & ~(size_t)255; return o; };
    size_t o_counts = alloc(64);
    size_t o_cursor = alloc(64);
    size_t o_segs   = alloc(2 * NEXP * 4);
    size_t o_task   = alloc((size_t)B_N * 4);
    size_t o_tke    = alloc((size_t)B_N * K_TOP * 4);
    size_t o_tkw    = alloc((size_t)B_N * K_TOP * 4);
    size_t o_slot   = alloc((size_t)B_N * K_TOP * 4);
    size_t o_Ag     = alloc((size_t)CAP * IN_DIM * 2);
    size_t o_act1   = alloc((size_t)CAP * W_DIM * 2);
    size_t o_act2   = alloc((size_t)CAP * W_DIM * 2);
    size_t o_out3   = alloc((size_t)CAP * W_DIM * 4);
    size_t o_fout   = alloc((size_t)B_N * W_DIM * 4);
    size_t o_Wp1    = alloc((size_t)NEXP * IN_DIM * W_DIM * 2);
    size_t o_Wp2    = alloc((size_t)NEXP * W_DIM * W_DIM * 2);
    size_t o_Wp3    = alloc((size_t)NEXP * W_DIM * W_DIM * 2);

    int*            counts = (int*)(ws + o_counts);
    int*            cursor = (int*)(ws + o_cursor);
    int*            segs   = (int*)(ws + o_segs);
    int*            task   = (int*)(ws + o_task);
    int*            tke    = (int*)(ws + o_tke);
    float*          tkw    = (float*)(ws + o_tkw);
    int*            slot   = (int*)(ws + o_slot);
    unsigned short* Ag     = (unsigned short*)(ws + o_Ag);
    unsigned short* act1   = (unsigned short*)(ws + o_act1);
    unsigned short* act2   = (unsigned short*)(ws + o_act2);
    float*          out3   = (float*)(ws + o_out3);
    float*          fout   = (float*)(ws + o_fout);
    unsigned short* Wp1    = (unsigned short*)(ws + o_Wp1);
    unsigned short* Wp2    = (unsigned short*)(ws + o_Wp2);
    unsigned short* Wp3    = (unsigned short*)(ws + o_Wp3);

    hipMemsetAsync(counts, 0, 64, stream);

    // routing
    gating_kernel<<<(B_N + 255) / 256, 256, 0, stream>>>(x, Wg, gbias, counts, task, tke, tkw);
    build_segs<<<1, 32, 0, stream>>>(counts, segs, cursor);
    scatter_gather<<<B_N, 256, 0, stream>>>(x, tke, cursor, slot, Ag);

    // weight repack (f32 -> bf16 fragment layout), once per call
    repack_w<<<NEXP * (IN_DIM / 32) * 8, 256, 0, stream>>>(We1, Ws1, Wp1, IN_DIM);
    repack_w<<<NEXP * (W_DIM / 32) * 8, 256, 0, stream>>>(We2, Ws2, Wp2, W_DIM);
    repack_w<<<NEXP * (W_DIM / 32) * 8, 256, 0, stream>>>(We3, Ws3, Wp3, W_DIM);

    // 3 WMMA GEMM layers over all segments (routed + shared)
    dim3 gg(CAP / 64, W_DIM / 256);
    moe_gemm<true,  false><<<gg, 256, 0, stream>>>(Ag,   IN_DIM, IN_DIM / 32, Wp1, be1, bs1, (void*)act1, segs);
    moe_gemm<true,  false><<<gg, 256, 0, stream>>>(act1, W_DIM,  W_DIM / 32,  Wp2, be2, bs2, (void*)act2, segs);
    moe_gemm<false, true ><<<gg, 256, 0, stream>>>(act2, W_DIM,  W_DIM / 32,  Wp3, be3, bs3, (void*)out3, segs);

    // combine + heads
    combine_kernel<<<dim3(B_N, W_DIM / 256), 256, 0, stream>>>(out3, slot, tkw, fout);
    heads_kernel<<<B_N, 256, 0, stream>>>(fout, Wh, bh, task, (float*)d_out);
}
